// LlamaAttention_65919158059110
// MI455X (gfx1250) — compile-verified, matrix operand order checked by reading, not run on hardware
//
#include <hip/hip_runtime.h>
#include <hip/hip_bf16.h>

// ---------------------------------------------------------------------------
// Llama attention layer for MI455X (gfx1250, wave32, WMMA 16x16x32 bf16,
// async global->LDS staging).
//   hidden [1,2048,2048] f32, w_qkv [2560,2048] f32, w_o [2048,2048] f32
//   out    [1,2048,2048] f32
// ---------------------------------------------------------------------------

#define S_LEN   2048
#define HID     2048
#define D_HEAD  64
#define NQH     32
#define NKVH    4
#define QKV_O   ((NQH + 2*NKVH) * D_HEAD)   // 2560
#define NHEADS_ALL (NQH + 2*NKVH)           // 40

typedef __attribute__((ext_vector_type(16))) __bf16 v16bf;
typedef __attribute__((ext_vector_type(8)))  __bf16 v8bf;
typedef __attribute__((ext_vector_type(8)))  float  v8f;

// Exact pointee type of the async-to-LDS builtin (from the compiler
// diagnostic): GCC-style vector_size(16) int, AS1 (global) / AS3 (LDS).
typedef __attribute__((__vector_size__(16))) int i32x4;
typedef __attribute__((address_space(1))) i32x4 as1_i32x4;
typedef __attribute__((address_space(3))) i32x4 as3_i32x4;

// ---------------------------------------------------------------------------
// gfx1250 async global->LDS copy (16B granules, ASYNCcnt-tracked) with a
// synchronous fallback if the toolchain lacks the builtin.
// ---------------------------------------------------------------------------
#if __has_builtin(__builtin_amdgcn_global_load_async_to_lds_b128)
#define HAS_ASYNC_LDS 1
#endif

__device__ __forceinline__ void cp16_async(const __bf16* g, __bf16* l) {
#ifdef HAS_ASYNC_LDS
    __builtin_amdgcn_global_load_async_to_lds_b128(
        (as1_i32x4*)(uintptr_t)g,
        (as3_i32x4*)(unsigned int)(uintptr_t)l,
        /*imm offset=*/0, /*cpol=*/0);
#else
    *(v8bf*)l = *(const v8bf*)g;
#endif
}

#if __has_builtin(__builtin_amdgcn_s_wait_asynccnt)
#define WAIT_ASYNC(n) __builtin_amdgcn_s_wait_asynccnt(n)
#else
#define WAIT_ASYNC(n) asm volatile("s_wait_asynccnt %0" :: "i"(n) : "memory")
#endif

// ---------------------------------------------------------------------------
// Fragment loader for 16-bit A/B operands (CDNA5 ISA 7.12.2):
//   lanes 0-15  : row = lane,    elems 0..7 -> K 0..7,   8..15 -> K 16..23
//   lanes 16-31 : row = lane-16, elems 0..7 -> K 8..15,  8..15 -> K 24..31
// base points at the 16-row subtile (row-major, row stride ld elements).
// ---------------------------------------------------------------------------
__device__ __forceinline__ v16bf load_frag16(const __bf16* base, int ld) {
    int lane = threadIdx.x & 31;
    int row  = lane & 15;
    int kb   = (lane >> 4) << 3;          // 0 or 8
    const __bf16* p = base + row * ld + kb;
    v8bf a0 = *(const v8bf*)(p);
    v8bf a1 = *(const v8bf*)(p + 16);
    return __builtin_shufflevector(a0, a1, 0,1,2,3,4,5,6,7,8,9,10,11,12,13,14,15);
}

__device__ __forceinline__ v8f wmma_bf16(v16bf a, v16bf b, v8f c) {
    return __builtin_amdgcn_wmma_f32_16x16x32_bf16(
        /*neg_a=*/false, a, /*neg_b=*/false, b,
        /*c_mod=*/(short)0, c, /*reuse_a=*/false, /*reuse_b=*/false);
}

// ---------------------------------------------------------------------------
// fp32 -> bf16 elementwise convert
// ---------------------------------------------------------------------------
__global__ void k_convert_bf16(const float* __restrict__ in,
                               __bf16* __restrict__ out, int n) {
    int i = blockIdx.x * blockDim.x + threadIdx.x;
    if (i < n) out[i] = (__bf16)in[i];
}

// ---------------------------------------------------------------------------
// bf16 GEMM:  C[M][N] = A[M][K] * B[N][K]^T   (both operands K-contiguous)
// Block: 256 threads (8 waves), tile 128x128, BK=32, double-buffered LDS
// filled with async global->LDS DMA.  Wave w computes M-subtiles
// {2*(w&3), 2*(w&3)+1} x N-subtiles {4*(w>>2)..+3}  (8 WMMA / K-step).
// ---------------------------------------------------------------------------
#define GBM 128
#define GBN 128
#define GBK 32
__global__ __launch_bounds__(256)
void k_gemm_bf16(const __bf16* __restrict__ A, const __bf16* __restrict__ B,
                 float* __restrict__ C, int M, int N, int K) {
    __shared__ __bf16 As[2][GBM][GBK];
    __shared__ __bf16 Bs[2][GBN][GBK];

    int tid  = threadIdx.x;
    int wave = tid >> 5;
    int lane = tid & 31;
    int m0 = blockIdx.x * GBM;
    int n0 = blockIdx.y * GBN;
    int wmp = (wave & 3) << 1;      // M subtile base (0,2,4,6)
    int wnp = (wave >> 2) << 2;     // N subtile base (0 or 4)

    v8f acc[2][4];
#pragma unroll
    for (int mi = 0; mi < 2; ++mi)
#pragma unroll
        for (int ni = 0; ni < 4; ++ni) acc[mi][ni] = (v8f){};

    // Stage loader: 128x32 bf16 for A and B each = 512 x b128; 256 threads
    // issue 2 A + 2 B async 16B copies per stage.
    auto issue_stage = [&](int k0, int buf) {
#pragma unroll
        for (int j = 0; j < 2; ++j) {
            int i   = (j << 8) + tid;       // 0..511
            int row = i >> 2;
            int c8  = (i & 3) << 3;
            cp16_async(&A[(size_t)(m0 + row) * K + k0 + c8], &As[buf][row][c8]);
            cp16_async(&B[(size_t)(n0 + row) * K + k0 + c8], &Bs[buf][row][c8]);
        }
    };

    int nk = K / GBK;
    issue_stage(0, 0);

    for (int kk = 0; kk < nk; ++kk) {
        int buf = kk & 1;
        if (kk + 1 < nk) {
            issue_stage((kk + 1) * GBK, buf ^ 1);
            WAIT_ASYNC(4);              // stage kk complete; kk+1 in flight
        } else {
            WAIT_ASYNC(0);
        }
        __syncthreads();

        v16bf af0 = load_frag16(&As[buf][wmp * 16][0], GBK);
        v16bf af1 = load_frag16(&As[buf][(wmp + 1) * 16][0], GBK);
        v16bf bfr[4];
#pragma unroll
        for (int ni = 0; ni < 4; ++ni)
            bfr[ni] = load_frag16(&Bs[buf][(wnp + ni) * 16][0], GBK);

#pragma unroll
        for (int ni = 0; ni < 4; ++ni) {
            acc[0][ni] = wmma_bf16(af0, bfr[ni], acc[0][ni]);
            acc[1][ni] = wmma_bf16(af1, bfr[ni], acc[1][ni]);
        }
        __syncthreads();                 // done reading buf before reuse
    }

    // C fragment layout: elem r of lane -> M = r + 8*(lane>>4), N = lane&15
    int coln = lane & 15;
#pragma unroll
    for (int mi = 0; mi < 2; ++mi) {
        int mrow = m0 + (wmp + mi) * 16 + 8 * (lane >> 4);
#pragma unroll
        for (int ni = 0; ni < 4; ++ni) {
            int nc = n0 + (wnp + ni) * 16 + coln;
#pragma unroll
            for (int r = 0; r < 8; ++r)
                C[(size_t)(mrow + r) * N + nc] = acc[mi][ni][r];
        }
    }
}

// ---------------------------------------------------------------------------
// RoPE + head split.  One thread per (s, head-of-40).
//   qkv [s][2560] f32  ->  Q [s][32][64] bf16 (roped)
//                          K [4][s][64]  bf16 (roped)
//                          V [4][s][64]  bf16
// ---------------------------------------------------------------------------
__global__ void k_rope_split(const float* __restrict__ qkv,
                             __bf16* __restrict__ Qb,
                             __bf16* __restrict__ Kb,
                             __bf16* __restrict__ Vb) {
    int idx = blockIdx.x * blockDim.x + threadIdx.x;
    if (idx >= S_LEN * NHEADS_ALL) return;
    int s  = idx / NHEADS_ALL;
    int hh = idx - s * NHEADS_ALL;
    const float* src = qkv + (size_t)s * QKV_O + hh * D_HEAD;

    if (hh < NQH + NKVH) {
        __bf16* dst = (hh < NQH)
            ? (Qb + (size_t)s * (NQH * D_HEAD) + hh * D_HEAD)
            : (Kb + ((size_t)(hh - NQH) * S_LEN + s) * D_HEAD);
#pragma unroll 8
        for (int i = 0; i < D_HEAD / 2; ++i) {
            float inv  = __powf(10000.0f, -(float)(2 * i) / (float)D_HEAD);
            float ang  = (float)s * inv;
            float c, sn;
            __sincosf(ang, &sn, &c);
            float x1 = src[i], x2 = src[i + D_HEAD / 2];
            dst[i]              = (__bf16)(x1 * c - x2 * sn);
            dst[i + D_HEAD / 2] = (__bf16)(x2 * c + x1 * sn);
        }
    } else {
        __bf16* dst = Vb + ((size_t)(hh - NQH - NKVH) * S_LEN + s) * D_HEAD;
#pragma unroll 8
        for (int i = 0; i < D_HEAD; ++i) dst[i] = (__bf16)src[i];
    }
}

// ---------------------------------------------------------------------------
// Flash attention (causal, GQA 8:1).
// Block = 128 threads (4 waves); blockIdx.x = q-tile of 64 rows, blockIdx.y = head.
// Each wave owns 16 q rows; Q fragments live in registers.  Per 64-key block:
// K tile staged via async global->LDS DMA, V transposed via VALU while the
// DMA is in flight, then 8 WMMA for S=QK^T and 8 WMMA for O+=P*V.
// ---------------------------------------------------------------------------
#define AQT 64      // q rows per block
#define AKT 64      // keys per block
__global__ __launch_bounds__(128)
void k_attn(const __bf16* __restrict__ Qb, const __bf16* __restrict__ Kb,
            const __bf16* __restrict__ Vb, __bf16* __restrict__ Ob) {
    __shared__ __bf16 Ks[AKT][D_HEAD];       // [key][dim]
    __shared__ __bf16 Vt[D_HEAD][AKT];       // [dim][key]  (transposed V)
    __shared__ __bf16 Ps[4][16][AKT];        // per-wave P scratch (C->A relayout)

    int h    = blockIdx.y;
    int kvh  = h >> 3;                        // 8 q heads per kv head
    int q0   = blockIdx.x * AQT;
    int tid  = threadIdx.x;
    int wave = tid >> 5;
    int lane = tid & 31;
    int qw   = q0 + wave * 16;                // this wave's q base row

    // Q fragments (dims 0..31 and 32..63) held in registers
    const __bf16* qbase = Qb + (size_t)qw * (NQH * D_HEAD) + h * D_HEAD;
    v16bf qa0 = load_frag16(qbase,      NQH * D_HEAD);
    v16bf qa1 = load_frag16(qbase + 32, NQH * D_HEAD);

    float m_r[8], l_r[8];
#pragma unroll
    for (int r = 0; r < 8; ++r) { m_r[r] = -1e30f; l_r[r] = 0.0f; }
    v8f o[4];
#pragma unroll
    for (int dn = 0; dn < 4; ++dn) o[dn] = (v8f){};

    const float scale = 0.125f;               // 1/sqrt(64)
    int qrow_base = qw + 8 * (lane >> 4);
    int coln      = lane & 15;

    for (int kb = 0; kb < q0 + AQT; kb += AKT) {
        const __bf16* ksrc = Kb + ((size_t)kvh * S_LEN + kb) * D_HEAD;
        const __bf16* vsrc = Vb + ((size_t)kvh * S_LEN + kb) * D_HEAD;

        // K tile: async DMA straight into LDS (no VGPR round-trip)
#pragma unroll
        for (int i = tid; i < AKT * 8; i += 128) {
            int key = i >> 3, c = (i & 7) << 3;
            cp16_async(&ksrc[(size_t)key * D_HEAD + c], &Ks[key][c]);
        }
        // V tile: transpose through VALU while K DMA is in flight
#pragma unroll
        for (int i = tid; i < AKT * 8; i += 128) {
            int key = i >> 3, c = (i & 7) << 3;
            v8bf vv = *(const v8bf*)&vsrc[(size_t)key * D_HEAD + c];
#pragma unroll
            for (int j = 0; j < 8; ++j) Vt[c + j][key] = vv[j];
        }
        WAIT_ASYNC(0);
        __syncthreads();

        // ---- S = Q * K^T  (4 key subtiles x 2 K-dim chunks) ----
        v8f sfr[4];
#pragma unroll
        for (int kn = 0; kn < 4; ++kn) {
            v16bf b0 = load_frag16(&Ks[kn * 16][0], D_HEAD);
            v16bf b1 = load_frag16(&Ks[kn * 16][32], D_HEAD);
            v8f a = (v8f){};
            a = wmma_bf16(qa0, b0, a);
            a = wmma_bf16(qa1, b1, a);
            sfr[kn] = a;
        }

        // ---- scale + causal mask ----
#pragma unroll
        for (int kn = 0; kn < 4; ++kn) {
#pragma unroll
            for (int r = 0; r < 8; ++r) {
                int qrow = qrow_base + r;
                int kcol = kb + kn * 16 + coln;
                sfr[kn][r] = (kcol <= qrow) ? sfr[kn][r] * scale : -3.0e38f;
            }
        }

        // ---- online softmax (row reductions across 16 lanes, width 16) ----
        float alpha[8];
#pragma unroll
        for (int r = 0; r < 8; ++r) {
            float v = sfr[0][r];
            v = fmaxf(v, sfr[1][r]); v = fmaxf(v, sfr[2][r]); v = fmaxf(v, sfr[3][r]);
            v = fmaxf(v, __shfl_xor(v, 1, 16));
            v = fmaxf(v, __shfl_xor(v, 2, 16));
            v = fmaxf(v, __shfl_xor(v, 4, 16));
            v = fmaxf(v, __shfl_xor(v, 8, 16));
            float mnew = fmaxf(m_r[r], v);
            alpha[r]   = __expf(m_r[r] - mnew);
            m_r[r]     = mnew;
            float rs = 0.0f;
#pragma unroll
            for (int kn = 0; kn < 4; ++kn) {
                float p = __expf(sfr[kn][r] - mnew);
                sfr[kn][r] = p;
                rs += p;
            }
            rs += __shfl_xor(rs, 1, 16);
            rs += __shfl_xor(rs, 2, 16);
            rs += __shfl_xor(rs, 4, 16);
            rs += __shfl_xor(rs, 8, 16);
            l_r[r] = l_r[r] * alpha[r] + rs;
        }
#pragma unroll
        for (int dn = 0; dn < 4; ++dn)
#pragma unroll
            for (int r = 0; r < 8; ++r) o[dn][r] *= alpha[r];

        // ---- relayout P (C-frag -> A-frag) through per-wave LDS scratch ----
        int prow = 8 * (lane >> 4);
#pragma unroll
        for (int kn = 0; kn < 4; ++kn)
#pragma unroll
            for (int r = 0; r < 8; ++r)
                Ps[wave][prow + r][kn * 16 + coln] = (__bf16)sfr[kn][r];

        v16bf pa0 = load_frag16(&Ps[wave][0][0], AKT);   // keys 0..31
        v16bf pa1 = load_frag16(&Ps[wave][0][32], AKT);  // keys 32..63

        // ---- O += P * V ----
#pragma unroll
        for (int dn = 0; dn < 4; ++dn) {
            v16bf vb0 = load_frag16(&Vt[dn * 16][0], AKT);
            v16bf vb1 = load_frag16(&Vt[dn * 16][32], AKT);
            o[dn] = wmma_bf16(pa0, vb0, o[dn]);
            o[dn] = wmma_bf16(pa1, vb1, o[dn]);
        }
        __syncthreads();
    }

    // ---- epilogue: O /= l, write bf16 [s][h*64+dim] ----
#pragma unroll
    for (int dn = 0; dn < 4; ++dn) {
#pragma unroll
        for (int r = 0; r < 8; ++r) {
            int qrow = qrow_base + r;
            int dim  = dn * 16 + coln;
            Ob[(size_t)qrow * (NQH * D_HEAD) + h * D_HEAD + dim] =
                (__bf16)(o[dn][r] / l_r[r]);
        }
    }
}

// ---------------------------------------------------------------------------
// Host-side launch
// ---------------------------------------------------------------------------
extern "C" void kernel_launch(void* const* d_in, const int* in_sizes, int n_in,
                              void* d_out, int out_size, void* d_ws, size_t ws_size,
                              hipStream_t stream) {
    const float* hidden = (const float*)d_in[0];   // [2048][2048]
    const float* w_qkv  = (const float*)d_in[1];   // [2560][2048]
    const float* w_o    = (const float*)d_in[2];   // [2048][2048]
    float* out = (float*)d_out;                    // [2048][2048]

    char* ws = (char*)d_ws;
    size_t off = 0;
    auto take = [&](size_t bytes) -> char* {
        char* p = ws + off;
        off += (bytes + 255) & ~(size_t)255;
        return p;
    };
    __bf16* hid_bf  = (__bf16*)take((size_t)S_LEN * HID * 2);
    __bf16* wqkv_bf = (__bf16*)take((size_t)QKV_O * HID * 2);
    __bf16* wo_bf   = (__bf16*)take((size_t)HID * HID * 2);
    float*  qkv_f   = (float*) take((size_t)S_LEN * QKV_O * 4);
    __bf16* Qb      = (__bf16*)take((size_t)S_LEN * NQH * D_HEAD * 2);
    __bf16* Kb      = (__bf16*)take((size_t)NKVH * S_LEN * D_HEAD * 2);
    __bf16* Vb      = (__bf16*)take((size_t)NKVH * S_LEN * D_HEAD * 2);
    __bf16* Ob      = (__bf16*)take((size_t)S_LEN * NQH * D_HEAD * 2);

    // 1) fp32 -> bf16 converts
    {
        int n1 = S_LEN * HID;
        int n2 = QKV_O * HID;
        k_convert_bf16<<<(n1 + 255) / 256, 256, 0, stream>>>(hidden, hid_bf, n1);
        k_convert_bf16<<<(n2 + 255) / 256, 256, 0, stream>>>(w_qkv, wqkv_bf, n2);
        k_convert_bf16<<<(n1 + 255) / 256, 256, 0, stream>>>(w_o, wo_bf, n1);
    }

    // 2) QKV projection: qkv[s][o] = hidden[s][:] . w_qkv[o][:]
    k_gemm_bf16<<<dim3(S_LEN / GBM, QKV_O / GBN), 256, 0, stream>>>(
        hid_bf, wqkv_bf, qkv_f, S_LEN, QKV_O, HID);

    // 3) RoPE + split into Q/K/V head layouts
    k_rope_split<<<(S_LEN * NHEADS_ALL + 255) / 256, 256, 0, stream>>>(
        qkv_f, Qb, Kb, Vb);

    // 4) causal flash attention
    k_attn<<<dim3(S_LEN / AQT, NQH), 128, 0, stream>>>(Qb, Kb, Vb, Ob);

    // 5) output projection: out[s][h] = Ob[s][:] . w_o[h][:]
    k_gemm_bf16<<<dim3(S_LEN / GBM, HID / GBN), 256, 0, stream>>>(
        Ob, wo_bf, out, S_LEN, HID, HID);
}
